// RESNET_INT_32409823215633
// MI455X (gfx1250) — 8-bit matrix operand order repaired and checked by reading, compiled, not run on hardware
//
#include <hip/hip_runtime.h>
#include <hip/hip_bf16.h>


typedef int v8i  __attribute__((ext_vector_type(8)));
typedef int gv4i __attribute__((vector_size(16)));

#define AS1 __attribute__((address_space(1)))
#define AS3 __attribute__((address_space(3)))

#define B_SZ 16

#if defined(__gfx1250__) &&                                                    \
    __has_builtin(__builtin_amdgcn_global_load_async_to_lds_b128) &&           \
    __has_builtin(__builtin_amdgcn_s_wait_asynccnt)
#define USE_ASYNC_LDS 1
#else
#define USE_ASYNC_LDS 0
#endif

__device__ __forceinline__ float wscale_from(const int* slot) {
    return __int_as_float(*slot) / 7.0f + 1e-8f;
}

__device__ __forceinline__ void stage16(const void* g, void* l, bool valid) {
#if USE_ASYNC_LDS
    if (valid) {
        __builtin_amdgcn_global_load_async_to_lds_b128(
            (AS1 gv4i*)g, (AS3 gv4i*)l, 0, 0);
    } else {
        *(int4*)l = make_int4(0, 0, 0, 0);
    }
#else
    int4 v = valid ? *(const int4*)g : make_int4(0, 0, 0, 0);
    *(int4*)l = v;
#endif
}

__device__ __forceinline__ void stage_fence() {
#if USE_ASYNC_LDS
    __builtin_amdgcn_s_wait_asynccnt(0);
#endif
    __syncthreads();
}


__global__ void absmax_kernel(const float* __restrict__ w, int n, int* __restrict__ slot) {
    float m = 0.0f;
    for (int i = blockIdx.x * blockDim.x + threadIdx.x; i < n; i += gridDim.x * blockDim.x)
        m = fmaxf(m, fabsf(w[i]));
    __shared__ float red[256];
    red[threadIdx.x] = m;
    __syncthreads();
    for (int s = 128; s > 0; s >>= 1) {
        if (threadIdx.x < s) red[threadIdx.x] = fmaxf(red[threadIdx.x], red[threadIdx.x + s]);
        __syncthreads();
    }
    if (threadIdx.x == 0) atomicMax(slot, __float_as_int(red[0]));
}

__global__ void quant_conv_w(const float* __restrict__ w, signed char* __restrict__ wq,
                             int O, int I, int KHW, const int* __restrict__ slot) {
    float s = wscale_from(slot);
    int K = KHW * I, n = O * K;
    for (int i = blockIdx.x * blockDim.x + threadIdx.x; i < n; i += gridDim.x * blockDim.x) {
        int o = i / K, k = i - o * K;
        int rs = k / I, c = k - rs * I;
        float q = rintf(w[(o * I + c) * KHW + rs] / s);
        wq[i] = (signed char)fminf(fmaxf(q, -8.0f), 7.0f);
    }
}

__global__ void quant_fc1_w(const float* __restrict__ w, signed char* __restrict__ wq,
                            const int* __restrict__ slot) {
    float s = wscale_from(slot);
    const int n = 1024 * 4096;
    for (int i = blockIdx.x * blockDim.x + threadIdx.x; i < n; i += gridDim.x * blockDim.x) {
        int o = i >> 12, kp = i & 4095;
        int sp = kp >> 8, c = kp & 255;
        float q = rintf(w[o * 4096 + c * 16 + sp] / s);
        wq[i] = (signed char)fminf(fmaxf(q, -8.0f), 7.0f);
    }
}

__global__ void stem_kernel(const float* __restrict__ x, const signed char* __restrict__ wq,
                            const int* __restrict__ slot,
                            const float* __restrict__ gamma, const float* __restrict__ beta,
                            const float* __restrict__ scale, unsigned char* __restrict__ out) {
    long long idx = (long long)blockIdx.x * blockDim.x + threadIdx.x;
    const long long total = (long long)B_SZ * 256 * 256 * 32;
    if (idx >= total) return;
    int co = (int)(idx & 31);
    long long p = idx >> 5;
    int xx = (int)(p & 255); p >>= 8;
    int yy = (int)(p & 255); p >>= 8;
    int b = (int)p;
    float ws = wscale_from(slot);
    float acc = 0.0f;
    for (int r = 0; r < 3; ++r) {
        int iy = yy - 1 + r;
        if (iy < 0 || iy > 255) continue;
        for (int s2 = 0; s2 < 3; ++s2) {
            int ix = xx - 1 + s2;
            if (ix < 0 || ix > 255) continue;
            for (int c = 0; c < 3; ++c) {
                float wv = (float)wq[co * 27 + (r * 3 + s2) * 3 + c] * ws;
                acc += wv * x[(((long long)b * 3 + c) * 256 + iy) * 256 + ix];
            }
        }
    }
    float y = gamma[co] * acc + beta[co];
    float q = fminf(fmaxf(rintf(y / *scale), 0.0f), 15.0f);
    out[idx] = (unsigned char)q;
}

__device__ __forceinline__ void epi_store(const v8i& acc, float deq, int mgbase,
                                          long long ng, long long npix, int Cout,
                                          const float* gamma, const float* beta,
                                          const float* qscale,
                                          unsigned char* qout, float* fout) {
    if (ng >= npix) return;
    if (gamma) {
        float qs = *qscale;
        unsigned long long pack = 0ull;
#pragma unroll
        for (int t = 0; t < 8; ++t) {
            float f = (float)acc[t] * deq;
            float y = gamma[mgbase + t] * f + beta[mgbase + t];
            unsigned long long q =
                (unsigned long long)fminf(fmaxf(rintf(y / qs), 0.0f), 15.0f);
            pack |= q << (8 * t);
        }
        *(unsigned long long*)(qout + ng * Cout + mgbase) = pack;
    } else {
        float* dst = fout + ng * Cout + mgbase;
        float4 v0, v1;
        v0.x = (float)acc[0] * deq; v0.y = (float)acc[1] * deq;
        v0.z = (float)acc[2] * deq; v0.w = (float)acc[3] * deq;
        v1.x = (float)acc[4] * deq; v1.y = (float)acc[5] * deq;
        v1.z = (float)acc[6] * deq; v1.w = (float)acc[7] * deq;
        *(float4*)dst = v0;
        *(float4*)(dst + 4) = v1;
    }
}

__global__ __launch_bounds__(256)
void conv_wmma_iu8(const unsigned char* __restrict__ act,
                   const signed char* __restrict__ wq,
                   const int* __restrict__ wslot, const float* __restrict__ ascale_in,
                   int Hi, int Wi, int Cin, int Cout,
                   int Ho, int Wo, int stride, int pad, int KHW,
                   int cinLog, int woLog, int pixLog,
                   const float* __restrict__ gamma, const float* __restrict__ beta,
                   const float* __restrict__ qscale,
                   unsigned char* __restrict__ qout,
                   float* __restrict__ fout) {
    __shared__ __align__(16) signed char   lA[32 * 64];
    __shared__ __align__(16) unsigned char lB[128 * 64];
    const int K = KHW * Cin;
    const int kcnt = (K + 63) >> 6;
    const long long npix = (long long)B_SZ * Ho * Wo;
    const long long ntile = (long long)blockIdx.x * 128;
    const int mtile = blockIdx.y * 32;
    const int tid = threadIdx.x;
    const int wv = tid >> 5, lane = tid & 31;
    const int n0 = wv << 4;
    const int half = lane >> 4, lm = lane & 15;
    const int cinMask = Cin - 1, woMask = (1 << woLog) - 1, pixMask = (1 << pixLog) - 1;
    v8i acc0 = {}, acc1 = {};
    for (int kc = 0; kc < kcnt; ++kc) {
        const int kbase = kc << 6;
        if (kc + 1 < kcnt && tid == 0)
            __builtin_prefetch(wq + (long long)mtile * K + kbase + 64, 0, 1);
        if (tid < 128) {
            int m = tid >> 2;
            int kg = kbase + ((tid & 3) << 4);
            stage16(wq + (long long)(mtile + m) * K + kg, lA + (tid << 4), kg < K);
        }
#pragma unroll
        for (int seg = 0; seg < 2; ++seg) {
            int i = tid + (seg << 8);
            int n = i >> 2;
            int kg = kbase + ((i & 3) << 4);
            long long ng = ntile + n;
            bool valid = false;
            const unsigned char* src = act;
            if (kg < K && ng < npix) {
                int rs = kg >> cinLog, c = kg & cinMask;
                int r = rs / 3, s2 = rs - r * 3;
                int b = (int)(ng >> pixLog);
                int rem = (int)ng & pixMask;
                int y = rem >> woLog, xx = rem & woMask;
                int iy = y * stride - pad + r, ix = xx * stride - pad + s2;
                if (iy >= 0 && iy < Hi && ix >= 0 && ix < Wi) {
                    valid = true;
                    src = act + ((long long)(b * Hi + iy) * Wi + ix) * Cin + c;
                }
            }
            stage16(src, lB + (i << 4), valid);
        }
        stage_fence();
        const int* pA = reinterpret_cast<const int*>(lA);
        const int* pB = reinterpret_cast<const int*>(lB);
        v8i a0, a1, bf;
#pragma unroll
        for (int t = 0; t < 8; ++t) {
            int ka = ((t >> 1) << 4) + (half << 3) + ((t & 1) << 2);
            a0[t] = pA[(lm << 4) + (ka >> 2)];
            a1[t] = pA[((16 + lm) << 4) + (ka >> 2)];
            int kb = ((t >> 1) << 4) + (half << 3) + ((t & 1) << 2);
            bf[t] = pB[((n0 + lm) << 4) + (kb >> 2)];
        }
        acc0 = __builtin_amdgcn_wmma_i32_16x16x64_iu8(true, a0, false, bf, acc0, false, false);
        acc1 = __builtin_amdgcn_wmma_i32_16x16x64_iu8(true, a1, false, bf, acc1, false, false);
        __syncthreads();
    }
    const float deq = wscale_from(wslot) * (*ascale_in);
    const int mb = (lane >> 4) * 8;
    const long long ng = ntile + n0 + lm;
    epi_store(acc0, deq, mtile + mb,      ng, npix, Cout, gamma, beta, qscale, qout, fout);
    epi_store(acc1, deq, mtile + 16 + mb, ng, npix, Cout, gamma, beta, qscale, qout, fout);
}

__global__ void add_sbn_quant4(const float4* __restrict__ r, const float4* __restrict__ s,
                               const float* __restrict__ gamma, const float* __restrict__ beta,
                               const float* __restrict__ scale, unsigned int* __restrict__ out,
                               long long n4, int cMask) {
    long long i = (long long)blockIdx.x * blockDim.x + threadIdx.x;
    if (i >= n4) return;
    int c = (int)((i << 2) & cMask);
    float4 rv = r[i], sv = s[i];
    float qs = *scale;
    float v[4] = {rv.x + sv.x, rv.y + sv.y, rv.z + sv.z, rv.w + sv.w};
    unsigned int pack = 0;
#pragma unroll
    for (int t = 0; t < 4; ++t) {
        float y = gamma[c + t] * v[t] + beta[c + t];
        unsigned int q = (unsigned int)fminf(fmaxf(rintf(y / qs), 0.0f), 15.0f);
        pack |= q << (8 * t);
    }
    out[i] = pack;
}

__global__ __launch_bounds__(256)
void gemm_wmma_iu8(const unsigned char* __restrict__ act,
                   const signed char* __restrict__ wq,
                   const int* __restrict__ wslot, const float* __restrict__ ascale_in,
                   int M, int K, int NB,
                   const float* __restrict__ gamma, const float* __restrict__ beta,
                   const float* __restrict__ qscale,
                   unsigned char* __restrict__ qout,
                   float* __restrict__ fout,
                   int signed_out) {
    __shared__ __align__(16) signed char   lA[128 * 64];
    __shared__ __align__(16) unsigned char lB[16 * 64];
    const int mtile = blockIdx.x * 128;
    const int tid = threadIdx.x, wv = tid >> 5, lane = tid & 31;
    const int m0 = wv << 4;
    const int half = lane >> 4, lm = lane & 15;
    const int kcnt = (K + 63) >> 6;
    v8i acc = {};
    for (int kc = 0; kc < kcnt; ++kc) {
        const int kbase = kc << 6;
#pragma unroll
        for (int seg = 0; seg < 2; ++seg) {
            int i = tid + (seg << 8);
            int m = i >> 2, mg = mtile + m;
            int kg = kbase + ((i & 3) << 4);
            stage16(wq + (long long)mg * K + kg, lA + (i << 4), mg < M && kg < K);
        }
        if (tid < 64) {
            int n = tid >> 2;
            int kg = kbase + ((tid & 3) << 4);
            stage16(act + (long long)n * K + kg, lB + (tid << 4), n < NB && kg < K);
        }
        stage_fence();
        const int* pA = reinterpret_cast<const int*>(lA);
        const int* pB = reinterpret_cast<const int*>(lB);
        v8i af, bf;
#pragma unroll
        for (int t = 0; t < 8; ++t) {
            int ka = ((t >> 1) << 4) + (half << 3) + ((t & 1) << 2);
            af[t] = pA[((m0 + lm) << 4) + (ka >> 2)];
            int kb = ((t >> 1) << 4) + (half << 3) + ((t & 1) << 2);
            bf[t] = pB[(lm << 4) + (kb >> 2)];
        }
        acc = __builtin_amdgcn_wmma_i32_16x16x64_iu8(true, af, false, bf, acc, false, false);
        __syncthreads();
    }
    const float deq = wscale_from(wslot) * (*ascale_in);
    const int ln = lane & 15, mb = (lane >> 4) * 8;
#pragma unroll
    for (int t = 0; t < 8; ++t) {
        int mg = mtile + m0 + mb + t;
        int n = ln;
        if (mg < M && n < NB) {
            float f = (float)acc[t] * deq;
            float y = gamma[mg] * f + beta[mg];
            float qs = *qscale;
            if (signed_out) {
                float q = fminf(fmaxf(rintf(y / qs), -8.0f), 7.0f) * qs;
                fout[(long long)n * M + mg] = q;
            } else {
                qout[(long long)n * M + mg] =
                    (unsigned char)fminf(fmaxf(rintf(y / qs), 0.0f), 15.0f);
            }
        }
    }
}


struct BlkCfg { int cin, cout, stride; };
static const BlkCfg BL[9] = {{32, 32, 2},  {32, 32, 1},   {32, 64, 2},
                             {64, 64, 1},  {64, 128, 2},  {128, 128, 1},
                             {128, 256, 2},{256, 256, 2}, {256, 256, 2}};

static inline int ilog2(int v) { int l = 0; while ((1 << l) < v) ++l; return l; }

extern "C" void kernel_launch(void* const* d_in, const int* in_sizes, int n_in,
                              void* d_out, int out_size, void* d_ws, size_t ws_size,
                              hipStream_t stream) {
    (void)in_sizes; (void)n_in; (void)out_size; (void)ws_size;

    auto F = [&](int i) { return (const float*)d_in[i]; };
    const float* fc1_w  = F(81);
    const float* fc2_w  = F(82);
    const float* sbn0_b = F(83); const float* sbn0_g = F(84); const float* sbn0_s = F(85);
    const float* sfc1_b = F(86); const float* sfc1_g = F(87); const float* sfc1_s = F(88);
    const float* sfc2_b = F(89); const float* sfc2_g = F(90); const float* sfc2_s = F(91);
    const float* stem_w = F(92);
    const float* x_in   = F(93);

    char* ws = (char*)d_ws;
    size_t off = 0;
    auto alloc = [&](size_t bytes) -> void* {
        void* p = ws + off;
        off = (off + bytes + 255) & ~(size_t)255;
        return p;
    };
    int* absmax = (int*)alloc(32 * sizeof(int));
    signed char* wq_stem = (signed char*)alloc(32 * 27);
    signed char* wq1[9]; signed char* wq2[9]; signed char* wqs[9];
    for (int i = 0; i < 9; ++i) {
        wq1[i] = (signed char*)alloc((size_t)BL[i].cout * BL[i].cin * 9);
        wq2[i] = (signed char*)alloc((size_t)BL[i].cout * BL[i].cout * 9);
        wqs[i] = (signed char*)alloc((size_t)BL[i].cout * BL[i].cin);
    }
    signed char* wq_fc1 = (signed char*)alloc((size_t)1024 * 4096);
    signed char* wq_fc2 = (signed char*)alloc((size_t)1000 * 1024);
    const size_t ACT = (size_t)B_SZ * 256 * 256 * 32;
    unsigned char* qa = (unsigned char*)alloc(ACT);
    unsigned char* qb = (unsigned char*)alloc(ACT);
    unsigned char* qt = (unsigned char*)alloc(ACT / 2);
    float* rbuf = (float*)alloc((size_t)B_SZ * 128 * 128 * 32 * 4);
    float* sbuf = (float*)alloc((size_t)B_SZ * 128 * 128 * 32 * 4);
    unsigned char* fcact = (unsigned char*)alloc((size_t)B_SZ * 1024);

    (void)hipMemsetAsync(absmax, 0, 32 * sizeof(int), stream);

    auto amax = [&](const float* w, size_t n, int slot) {
        int blocks = (int)((n + 255) / 256); if (blocks > 1024) blocks = 1024;
        absmax_kernel<<<blocks, 256, 0, stream>>>(w, (int)n, absmax + slot);
    };
    auto qconv = [&](const float* w, signed char* wq, int O, int I, int KHW, int slot) {
        size_t n = (size_t)O * I * KHW;
        int blocks = (int)((n + 255) / 256); if (blocks > 4096) blocks = 4096;
        quant_conv_w<<<blocks, 256, 0, stream>>>(w, wq, O, I, KHW, absmax + slot);
    };

    amax(stem_w, 32 * 27, 0);
    for (int i = 0; i < 9; ++i) {
        amax(F(i * 9 + 6), (size_t)BL[i].cout * BL[i].cin * 9, 1 + 3 * i);
        amax(F(i * 9 + 7), (size_t)BL[i].cout * BL[i].cout * 9, 2 + 3 * i);
        amax(F(i * 9 + 8), (size_t)BL[i].cout * BL[i].cin, 3 + 3 * i);
    }
    amax(fc1_w, (size_t)1024 * 4096, 28);
    amax(fc2_w, (size_t)1000 * 1024, 29);

    qconv(stem_w, wq_stem, 32, 3, 9, 0);
    for (int i = 0; i < 9; ++i) {
        qconv(F(i * 9 + 6), wq1[i], BL[i].cout, BL[i].cin, 9, 1 + 3 * i);
        qconv(F(i * 9 + 7), wq2[i], BL[i].cout, BL[i].cout, 9, 2 + 3 * i);
        qconv(F(i * 9 + 8), wqs[i], BL[i].cout, BL[i].cin, 1, 3 + 3 * i);
    }
    quant_fc1_w<<<4096, 256, 0, stream>>>(fc1_w, wq_fc1, absmax + 28);
    qconv(fc2_w, wq_fc2, 1000, 1024, 1, 29);

    {
        long long total = (long long)B_SZ * 256 * 256 * 32;
        stem_kernel<<<(unsigned)((total + 255) / 256), 256, 0, stream>>>(
            x_in, wq_stem, absmax + 0, sbn0_g, sbn0_b, sbn0_s, qa);
    }

    int Hi = 256, Wi = 256;
    const float* prev_scale = sbn0_s;
    for (int i = 0; i < 9; ++i) {
        int cin = BL[i].cin, cout = BL[i].cout, st = BL[i].stride;
        int Ho = (Hi - 1) / st + 1, Wo = (Wi - 1) / st + 1;
        long long npix = (long long)B_SZ * Ho * Wo;
        dim3 grid((unsigned)((npix + 127) / 128), (unsigned)(cout / 32));
        const float* s1b = F(i * 9 + 0); const float* s1g = F(i * 9 + 1); const float* s1s = F(i * 9 + 2);
        const float* s2b = F(i * 9 + 3); const float* s2g = F(i * 9 + 4); const float* s2s = F(i * 9 + 5);
        int ciL = ilog2(cin), coL = ilog2(cout);
        int woL = ilog2(Wo), pxL = ilog2(Ho * Wo);
        conv_wmma_iu8<<<grid, 256, 0, stream>>>(qa, wq1[i], absmax + 1 + 3 * i, prev_scale,
            Hi, Wi, cin, cout, Ho, Wo, st, 1, 9, ciL, woL, pxL,
            s1g, s1b, s1s, qt, nullptr);
        conv_wmma_iu8<<<grid, 256, 0, stream>>>(qt, wq2[i], absmax + 2 + 3 * i, s1s,
            Ho, Wo, cout, cout, Ho, Wo, 1, 1, 9, coL, woL, pxL,
            nullptr, nullptr, nullptr, nullptr, rbuf);
        conv_wmma_iu8<<<grid, 256, 0, stream>>>(qa, wqs[i], absmax + 3 + 3 * i, prev_scale,
            Hi, Wi, cin, cout, Ho, Wo, st, 0, 1, ciL, woL, pxL,
            nullptr, nullptr, nullptr, nullptr, sbuf);
        long long n4 = npix * cout / 4;
        add_sbn_quant4<<<(unsigned)((n4 + 255) / 256), 256, 0, stream>>>(
            (const float4*)rbuf, (const float4*)sbuf, s2g, s2b, s2s,
            (unsigned int*)qb, n4, cout - 1);
        { unsigned char* t = qa; qa = qb; qb = t; }
        prev_scale = s2s;
        Hi = Ho; Wi = Wo;
    }

    gemm_wmma_iu8<<<(1024 + 127) / 128, 256, 0, stream>>>(
        qa, wq_fc1, absmax + 28, prev_scale, 1024, 4096, B_SZ,
        sfc1_g, sfc1_b, sfc1_s, fcact, nullptr, 0);

    gemm_wmma_iu8<<<(1000 + 127) / 128, 256, 0, stream>>>(
        fcact, wq_fc2, absmax + 29, sfc1_s, 1000, 1024, B_SZ,
        sfc2_g, sfc2_b, sfc2_s, nullptr, (float*)d_out, 1);
}
